// model_benchmark_41686952575154
// MI455X (gfx1250) — matrix operand order repaired and checked by reading, not compiled, not run on hardware
//
#include <hip/hip_runtime.h>

// ---------------------------------------------------------------------------
// GSN graph net for MI455X (gfx1250, wave32). One batch element per wave.
// All affine layers and per-element graph matmuls via v_wmma_f32_16x16x32_f16.
// Activations live in per-wave LDS tiles [32 rows x 32 cols] of f16 (2 KB),
// zero-padded so K-padding multiplies against zero weights.
// Adjacency tiles are staged global->LDS with the gfx1250 async path
// (global_load_async_to_lds_b32 + s_wait_asynccnt), overlapping with compute.
// ---------------------------------------------------------------------------

typedef __attribute__((ext_vector_type(16))) _Float16 v16h;
typedef __attribute__((ext_vector_type(8)))  float    v8f;
typedef __attribute__((ext_vector_type(4)))  unsigned int v4u;

#if defined(__gfx1250__) && \
    __has_builtin(__builtin_amdgcn_global_load_async_to_lds_b32) && \
    __has_builtin(__builtin_amdgcn_s_wait_asynccnt)
#define USE_ASYNC 1
#else
#define USE_ASYNC 0
#endif

union F16Frag {
    v16h     v;
    _Float16 h[16];
    v4u      q[2];
};

struct Params {
    const float* in;        // [B,4300]
    const float* w[19];     // sorted-key param order (incl. unused g1..g3)
    const float* b[19];
    float*       out;       // [B,20]
    int          nB;
};

#define NEG_SLOPE 0.01f

// LDS layout (dynamic shared):
//   [0      , 32768)  : 16 weight slots, each W^T as 32x32 f16 (2 KB)
//   [32768  , 34816)  : 16 bias slots, 32 f32 each (zero padded)
//   [34816  , ...  )  : per-wave region: 6 x 2 KB f16 tiles + 2 KB f32 staging
#define W_BYTES   32768
#define B_BYTES   2048
#define WAVE_BASE 34816
#define WAVE_SZ   14336
#define SMEM_TOTAL (WAVE_BASE + 8 * WAVE_SZ)   // 149,504 B (<160 KB -> 2 WG/WGP)

// Weight slots (order of used layers)
enum { S_DAG1 = 0, S_DAG2, S_DAG3, S_F1, S_F2, S_F3, S_FC1, S_FC2, S_FC3,
       S_FC4, S_GLOB1, S_GLOB2, S_GLOB3, S_H1, S_H2, S_H3 };

// ---- WMMA fragment loaders -------------------------------------------------

// A fragment (16x32 f16, M-tile `mtile`) from row-major LDS tile (stride 32 halfs).
// Layout: lanes 0-15 hold M=lane, K=0..15; lanes 16-31: K=16..31 (contiguous halves).
// Per lane: two ds_load_b128 at row*64 + hi*32 bytes and +16 bytes.
__device__ inline v16h load_afrag(const _Float16* act, int mtile, int lane) {
    const int m = lane & 15, hi = lane >> 4;
    const _Float16* p = act + (mtile * 16 + m) * 32 + hi * 16;
    F16Frag f;
    f.q[0] = *(const v4u*)(p);
    f.q[1] = *(const v4u*)(p + 8);
    return f.v;
}

// B fragment (32x16 f16, N-tile `ntile`) from W^T stored [n rows x 32 k] halfs.
// Lane n holds column n, K=0..15; lane n+16 holds K=16..31 (contiguous halves).
__device__ inline v16h load_bfrag_w(const _Float16* wT, int ntile, int lane) {
    const int n = lane & 15, hi = lane >> 4;
    const _Float16* p = wT + (ntile * 16 + n) * 32 + hi * 16;
    F16Frag f;
    f.q[0] = *(const v4u*)(p);
    f.q[1] = *(const v4u*)(p + 8);
    return f.v;
}

// B fragment from a row-major activation tile: column gather (16 x ds_load_u16).
__device__ inline v16h load_bfrag_act(const _Float16* act, int lane) {
    const int n = lane & 15, hi = lane >> 4;
    F16Frag f;
#pragma unroll
    for (int e = 0; e < 16; ++e) f.h[e] = act[(hi * 16 + e) * 32 + n];
    return f.v;
}

// ---- GraphLayer: dst = act(src @ W + b) ------------------------------------
template <bool RELU>
__device__ inline void layer(const _Float16* src, _Float16* dst,
                             const _Float16* wT, const float* bias,
                             int ntiles, int mtiles, int lane) {
    const int n = lane & 15, hi = lane >> 4;
#pragma unroll
    for (int nt = 0; nt < ntiles; ++nt) {
        v16h  bf = load_bfrag_w(wT, nt, lane);
        float bv = bias[nt * 16 + n];
#pragma unroll
        for (int mt = 0; mt < mtiles; ++mt) {
            v16h af  = load_afrag(src, mt, lane);
            v8f  acc = {};
            acc = __builtin_amdgcn_wmma_f32_16x16x32_f16(
                false, af, false, bf, (short)0, acc, false, false);
#pragma unroll
            for (int r = 0; r < 8; ++r) {
                float v = acc[r] + bv;
                if (RELU) v = (v >= 0.0f) ? v : NEG_SLOPE * v;
                dst[(mt * 16 + hi * 8 + r) * 32 + nt * 16 + n] = (_Float16)v;
            }
        }
    }
}

// ---- Per-element graph matmul: dst = G @ act (einsum 'nm,md->nd') ----------
__device__ inline void gmat(const _Float16* G, const _Float16* act,
                            _Float16* dst, int mtiles, int lane) {
    const int n = lane & 15, hi = lane >> 4;
    v16h bf = load_bfrag_act(act, lane);
#pragma unroll
    for (int mt = 0; mt < mtiles; ++mt) {
        v16h af  = load_afrag(G, mt, lane);
        v8f  acc = {};
        acc = __builtin_amdgcn_wmma_f32_16x16x32_f16(
            false, af, false, bf, (short)0, acc, false, false);
#pragma unroll
        for (int r = 0; r < 8; ++r)
            dst[(mt * 16 + hi * 8 + r) * 32 + n] = (_Float16)acc[r];
    }
}

// ---- Split global->LDS staging of a [rows x 20] f32 matrix -----------------
// stage_issue: start moving n raw f32 into the per-wave staging buffer.
// stage_finish: wait for it and convert into the zero-padded 32x32 f16 G tile.
__device__ inline void stage_issue(float* gf32, const float* src, int n, int lane) {
#if USE_ASYNC
    for (int idx = lane; idx < n; idx += 32) {
        int* gg = (int*)(src + idx);   // generic; cast away const/type first
        int* ll = (int*)(gf32 + idx);
        auto gp = (__attribute__((address_space(1))) int*)gg;
        auto lp = (__attribute__((address_space(3))) int*)ll;
        __builtin_amdgcn_global_load_async_to_lds_b32(gp, lp, 0, 0);
    }
#else
    for (int idx = lane; idx < n; idx += 32)
        gf32[idx] = src[idx];
#endif
}

__device__ inline void stage_finish(_Float16* G, const float* gf32, int rows, int lane) {
#if USE_ASYNC
    __builtin_amdgcn_s_wait_asynccnt(0);
#endif
    for (int idx = lane; idx < 400; idx += 32) {
        int r = idx / 20, c = idx % 20;
        float v = (r < rows) ? gf32[r * 20 + c] : 0.0f;
        G[r * 32 + c] = (_Float16)v;
    }
}

__global__ __launch_bounds__(256) void gsn_wmma_kernel(Params P) {
    extern __shared__ char smem[];
    _Float16* wbase = (_Float16*)smem;
    float*    bbase = (float*)(smem + W_BYTES);

    // ---- stage weights (block cooperative) ---------------------------------
    {
        uint32_t* z = (uint32_t*)smem;
        for (int i = threadIdx.x; i < (W_BYTES + B_BYTES) / 4; i += blockDim.x)
            z[i] = 0;
    }
    __syncthreads();
    {
        const int dims[19][2] = {
            {13,16},{16, 8},{ 8, 8},            // dag1..3
            { 8,16},{16, 8},{ 8, 8},            // f1..3
            {29,32},{32,16},{16, 8},{ 8, 1},    // fc1..4
            { 8,16},{16, 8},{ 8, 8},            // g1..3 (unused)
            { 8,16},{16, 8},{ 8, 8},            // glob1..3
            { 5,16},{16, 8},{ 8, 8}};           // h1..3
        const int slot[19] = {0,1,2, 3,4,5, 6,7,8,9, -1,-1,-1, 10,11,12, 13,14,15};
        for (int j = 0; j < 19; ++j) {
            int s = slot[j];
            if (s < 0) continue;
            int K = dims[j][0], N = dims[j][1];
            const float* w = P.w[j];
            for (int idx = threadIdx.x; idx < K * N; idx += blockDim.x) {
                int k = idx / N, c = idx % N;
                wbase[s * 1024 + c * 32 + k] = (_Float16)w[idx];   // W^T
            }
            const float* b = P.b[j];
            for (int idx = threadIdx.x; idx < N; idx += blockDim.x)
                bbase[s * 32 + idx] = b[idx];
        }
    }
    __syncthreads();

    const int wave = threadIdx.x >> 5;
    const int lane = threadIdx.x & 31;
    const int bidx = blockIdx.x * 8 + wave;
    if (bidx >= P.nB) return;   // uniform per wave: EXEC stays all-ones for WMMA

    char* wv = smem + WAVE_BASE + wave * WAVE_SZ;
    _Float16* BUF0 = (_Float16*)(wv + 0);
    _Float16* BUF1 = (_Float16*)(wv + 2048);
    _Float16* GM   = (_Float16*)(wv + 4096);
    _Float16* X    = (_Float16*)(wv + 6144);
    _Float16* NI   = (_Float16*)(wv + 8192);
    _Float16* DSM  = (_Float16*)(wv + 10240);
    float*    GF32 = (float*)(wv + 12288);     // raw f32 staging (async dest)

    for (int i = lane; i < WAVE_SZ / 4; i += 32) ((uint32_t*)wv)[i] = 0;

#define WT(s) (wbase + (s) * 1024)
#define BI(s) (bbase + (s) * 32)

    const float* row = P.in + (size_t)bidx * 4300;

    // node_inputs [20x5] -> NI
    for (int idx = lane; idx < 100; idx += 32)
        NI[(idx / 5) * 32 + (idx % 5)] = (_Float16)row[idx];

    // start fetching depth-0 adjacency while the h block runs
    stage_issue(GF32, row + 120, 400, lane);

    // h block: NI -> X
    layer<true>(NI,   BUF0, WT(S_H1), BI(S_H1), 1, 2, lane);
    layer<true>(BUF0, BUF1, WT(S_H2), BI(S_H2), 1, 2, lane);
    layer<true>(BUF1, X,    WT(S_H3), BI(S_H3), 1, 2, lane);

    // 8 message-passing depths
#pragma unroll 1
    for (int d = 0; d < 8; ++d) {
        layer<true>(X,    BUF0, WT(S_F1), BI(S_F1), 1, 2, lane);
        layer<true>(BUF0, BUF1, WT(S_F2), BI(S_F2), 1, 2, lane);
        layer<true>(BUF1, BUF0, WT(S_F3), BI(S_F3), 1, 2, lane);

        stage_finish(GM, GF32, 20, lane);               // gcn_mats[:,d]
        gmat(GM, BUF0, BUF1, 2, lane);                  // y = G @ y

        if (d < 7)                                      // overlap next fetch
            stage_issue(GF32, row + 120 + (d + 1) * 400, 400, lane);
        else
            stage_issue(GF32, row + 3480, 400, lane);   // summ_mats

        layer<true>(BUF1, BUF0, WT(S_F1), BI(S_F1), 1, 2, lane);
        layer<true>(BUF0, BUF1, WT(S_F2), BI(S_F2), 1, 2, lane);
        layer<true>(BUF1, BUF0, WT(S_F3), BI(S_F3), 1, 2, lane);

        // x += y * gcn_masks[:,d]
        const float* mk = row + 3320 + d * 20;
        for (int idx = lane; idx < 160; idx += 32) {
            int m = idx >> 3, n = idx & 7;
            float xv = (float)X[m * 32 + n] + (float)BUF0[m * 32 + n] * mk[m];
            X[m * 32 + n] = (_Float16)xv;
        }
    }

    // dag input concat [NI | X] (20x13) -> BUF0
    for (int idx = lane; idx < 260; idx += 32) {
        int m = idx / 13, c = idx % 13;
        BUF0[m * 32 + c] = (c < 5) ? NI[m * 32 + c] : X[m * 32 + (c - 5)];
    }
    layer<true>(BUF0, BUF1, WT(S_DAG1), BI(S_DAG1), 1, 2, lane);
    layer<true>(BUF1, BUF0, WT(S_DAG2), BI(S_DAG2), 1, 2, lane);
    layer<true>(BUF0, BUF1, WT(S_DAG3), BI(S_DAG3), 1, 2, lane);

    stage_finish(GM, GF32, 20, lane);           // summ_mats
    gmat(GM, BUF1, DSM, 2, lane);               // dag summary -> DSM

    stage_issue(GF32, row + 3880, 20, lane);    // running_dags_mat [1x20]

    layer<true>(DSM,  BUF0, WT(S_GLOB1), BI(S_GLOB1), 1, 2, lane);
    layer<true>(BUF0, BUF1, WT(S_GLOB2), BI(S_GLOB2), 1, 2, lane);
    layer<true>(BUF1, BUF0, WT(S_GLOB3), BI(S_GLOB3), 1, 2, lane);

    stage_finish(GM, GF32, 1, lane);
    gmat(GM, BUF0, BUF1, 1, lane);              // global summary -> BUF1 row 0

    stage_issue(GF32, row + 3900, 400, lane);   // dag_summ_backward_map
    stage_finish(GM, GF32, 20, lane);
    gmat(GM, DSM, BUF0, 2, lane);               // dag summary extend -> BUF0

    // merge concat [NI | X | extend | glob-broadcast] (20x29) -> GM
    for (int idx = lane; idx < 580; idx += 32) {
        int m = idx / 29, c = idx % 29;
        _Float16 v;
        if      (c < 5)  v = NI[m * 32 + c];
        else if (c < 13) v = X[m * 32 + (c - 5)];
        else if (c < 21) v = BUF0[m * 32 + (c - 13)];
        else             v = BUF1[c - 21];          // row 0 broadcast
        GM[m * 32 + c] = v;
    }

    // fc head
    layer<true >(GM,   BUF0, WT(S_FC1), BI(S_FC1), 2, 2, lane);  // 29->32
    layer<true >(BUF0, BUF1, WT(S_FC2), BI(S_FC2), 1, 2, lane);  // 32->16
    layer<true >(BUF1, BUF0, WT(S_FC3), BI(S_FC3), 1, 2, lane);  // 16->8
    layer<false>(BUF0, BUF1, WT(S_FC4), BI(S_FC4), 1, 2, lane);  // 8->1

    // output + validity mask
    if (lane < 20) {
        float v = (float)BUF1[lane * 32] + (row[100 + lane] - 1.0f) * 10000.0f;
        P.out[(size_t)bidx * 20 + lane] = v;
    }
#undef WT
#undef BI
}

extern "C" void kernel_launch(void* const* d_in, const int* in_sizes, int n_in,
                              void* d_out, int out_size, void* d_ws, size_t ws_size,
                              hipStream_t stream) {
    (void)d_ws; (void)ws_size; (void)out_size; (void)n_in;
    Params P;
    P.in = (const float*)d_in[0];
    for (int j = 0; j < 19; ++j) {
        P.w[j] = (const float*)d_in[1 + 2 * j];
        P.b[j] = (const float*)d_in[2 + 2 * j];
    }
    P.out = (float*)d_out;
    P.nB  = in_sizes[0] / 4300;

    const size_t shmem = SMEM_TOTAL;
    (void)hipFuncSetAttribute((const void*)gsn_wmma_kernel,
                              hipFuncAttributeMaxDynamicSharedMemorySize,
                              (int)shmem);
    int blocks = (P.nB + 7) / 8;       // 8 waves per block, 1 element per wave
    gsn_wmma_kernel<<<blocks, 256, shmem, stream>>>(P);
}
